// TRAMModel_66932770341356
// MI455X (gfx1250) — matrix operand order repaired and checked by reading, compiled, not run on hardware
//
#include <hip/hip_runtime.h>
#include <hip/hip_bf16.h>

// Problem constants (match reference)
constexpr int Bc = 128;   // batch
constexpr int Tc = 2048;  // time
constexpr int Ec = 384;   // embedding
constexpr int Hc = 128;   // hidden
constexpr int G3 = 3 * Hc; // 384 = gate width
constexpr int LDW = 40;    // padded LDS row width for W_ih staging (bank-conflict-free)
constexpr int LDH = Hc + 8; // padded LDS row width for hidden state staging

typedef __bf16 bf16_t;
typedef __bf16 v16bf __attribute__((ext_vector_type(16)));
typedef __bf16 bf8v  __attribute__((ext_vector_type(8)));
typedef float  v8f   __attribute__((ext_vector_type(8)));
typedef float  f4v   __attribute__((ext_vector_type(4)));

static __device__ __forceinline__ v8f vzero8() {
  v8f z = {0.f, 0.f, 0.f, 0.f, 0.f, 0.f, 0.f, 0.f};
  return z;
}

static __device__ __forceinline__ v8f wmma_bf16(v16bf a, v16bf b, v8f c) {
  // D = A(16x32 bf16) * B(32x16 bf16) + C(16x16 f32)
  return __builtin_amdgcn_wmma_f32_16x16x32_bf16(
      /*neg_a=*/false, a, /*neg_b=*/false, b,
      /*c_mod=*/(short)0, c, /*reuse_a=*/false, /*reuse_b=*/false);
}

static __device__ __forceinline__ float sigmoidf_(float x) {
  return 1.f / (1.f + __expf(-x));
}

// ---- CDNA5 async global->LDS copy (ASYNCcnt domain) ----------------------
static __device__ __forceinline__ void async_copy16(unsigned int lds_off,
                                                    const void* gaddr) {
  asm volatile("global_load_async_to_lds_b128 %0, %1, off"
               :: "v"(lds_off), "v"((unsigned long long)(uintptr_t)gaddr)
               : "memory");
}
static __device__ __forceinline__ void async_wait0() {
  asm volatile("s_wait_asynccnt 0x0" ::: "memory");
}

// ---- Fragment loaders per CDNA5 ISA 7.12.2 (wave32) ----------------------
// A 16x32 bf16: lane&15 = M row; lane>=16 holds K {8..15, 24..31}, else {0..7, 16..23}
static __device__ __forceinline__ v16bf frag_a_bf16(const bf16_t* p, int ld,
                                                    int lane, int k0) {
  int m = lane & 15, ks = lane >> 4;
  const bf16_t* r = p + m * ld + k0;
  bf8v lo = *(const bf8v*)(r + ks * 8);        // K = ks*8 .. +7
  bf8v hi = *(const bf8v*)(r + 16 + ks * 8);   // K = 16 + ks*8 .. +7
  v16bf a;
#pragma unroll
  for (int e = 0; e < 8; ++e) { a[e] = lo[e]; a[8 + e] = hi[e]; }
  return a;
}

// Same A layout but source is fp32 (convert in registers)
static __device__ __forceinline__ v16bf frag_a_f32(const float* p, int ld,
                                                   int lane, int k0) {
  int m = lane & 15, ks = lane >> 4;
  const float* r = p + m * ld + k0;
  f4v a0 = *(const f4v*)(r + ks * 8);
  f4v a1 = *(const f4v*)(r + ks * 8 + 4);
  f4v b0 = *(const f4v*)(r + 16 + ks * 8);
  f4v b1 = *(const f4v*)(r + 16 + ks * 8 + 4);
  v16bf a;
#pragma unroll
  for (int e = 0; e < 4; ++e) {
    a[e]      = (bf16_t)a0[e];
    a[4 + e]  = (bf16_t)a1[e];
    a[8 + e]  = (bf16_t)b0[e];
    a[12 + e] = (bf16_t)b1[e];
  }
  return a;
}

// B 32x16 bf16: lane&15 = N column; lane>=16 holds K {8..15, 24..31}, else {0..7, 16..23}.
// Source W[n][k] row-major (stride ld) so K is contiguous per lane.
static __device__ __forceinline__ v16bf frag_b_bf16(const bf16_t* p, int ld,
                                                    int lane, int k0) {
  int n = lane & 15, ks = lane >> 4;
  const bf16_t* r = p + n * ld + k0;
  bf8v lo = *(const bf8v*)(r + ks * 8);
  bf8v hi = *(const bf8v*)(r + 16 + ks * 8);
  v16bf b;
#pragma unroll
  for (int e = 0; e < 8; ++e) { b[e] = lo[e]; b[8 + e] = hi[e]; }
  return b;
}

// ---- Kernel 0: convert weights fp32 -> bf16 ------------------------------
__global__ __launch_bounds__(256) void k_convert(
    const float* __restrict__ Wih, const float* __restrict__ Whh,
    const float* __restrict__ W1, bf16_t* __restrict__ oWih,
    bf16_t* __restrict__ oWhh, bf16_t* __restrict__ oW1) {
  int i = blockIdx.x * 256 + threadIdx.x;
  if (i < G3 * Ec)  oWih[i] = (bf16_t)Wih[i];   // 384*384
  if (i < G3 * Hc)  oWhh[i] = (bf16_t)Whh[i];   // 384*128
  if (i < 64 * Hc)  oW1[i]  = (bf16_t)W1[i];    // 64*128
}

// ---- Kernel 1: x_proj = embs @ W_ih^T + b_ih  -> bf16 [B*T, 3H] ----------
// grid (BT/64), block 128 (4 waves). WG = 64 M rows x ALL 384 N cols so the
// 402MB embs tensor streams through exactly once (true BW-bound: ~0.6GB total).
// W_ih k-blocks double-buffer staged to LDS with async-to-LDS DMA; B-frags
// register double-buffered so ds_load of tile nt+1 overlaps WMMA of tile nt.
__global__ __launch_bounds__(128) void k_xproj(
    const float* __restrict__ embs, const bf16_t* __restrict__ Wih,
    const float* __restrict__ bih, bf16_t* __restrict__ xproj) {
  __shared__ bf16_t wls[2][G3 * LDW];  // 2 x 30KB, rows padded to 80B
  int tid = threadIdx.x;
  int lane = tid & 31;
  int w = tid >> 5;
  int m_base = blockIdx.x * 64 + w * 16;   // row in [0, B*T)

  v8f acc[24];
#pragma unroll
  for (int i = 0; i < 24; ++i) acc[i] = vzero8();

  // stage k-block 0 into buffer 0
  {
    unsigned int base = (unsigned int)(uintptr_t)&wls[0][0];
#pragma unroll
    for (int idx = tid; idx < G3 * 4; idx += 128) {     // 1536 x 16B chunks
      int n = idx >> 2, c = (idx & 3) * 8;
      async_copy16(base + (unsigned int)(n * LDW + c) * 2, Wih + n * Ec + c);
    }
  }
  async_wait0();
  __syncthreads();

  const float* arow = embs + (size_t)m_base * Ec;
  int cur = 0;
  for (int k0 = 0; k0 < Ec; k0 += 32) {
    if (k0 + 32 < Ec) {  // stage next k-block into the other buffer (async)
      unsigned int base = (unsigned int)(uintptr_t)&wls[cur ^ 1][0];
#pragma unroll
      for (int idx = tid; idx < G3 * 4; idx += 128) {
        int n = idx >> 2, c = (idx & 3) * 8;
        async_copy16(base + (unsigned int)(n * LDW + c) * 2,
                     Wih + n * Ec + (k0 + 32) + c);
      }
    }
    v16bf af = frag_a_f32(arow, Ec, lane, k0);
    __builtin_prefetch(arow + k0 + 64 + (lane & 15) * Ec, 0, 3);
    // register-double-buffered B fragments: keep one ds_load in flight
    v16bf bcur = frag_b_bf16(&wls[cur][0], LDW, lane, 0);
#pragma unroll
    for (int nt = 0; nt < 24; ++nt) {
      v16bf bnxt = bcur;
      if (nt < 23)
        bnxt = frag_b_bf16(&wls[cur][(nt + 1) * 16 * LDW], LDW, lane, 0);
      acc[nt] = wmma_bf16(af, bcur, acc[nt]);
      bcur = bnxt;
    }
    async_wait0();     // next buffer landed
    __syncthreads();   // everyone done reading cur before it becomes the target
    cur ^= 1;
  }

  int nloc = lane & 15;
  int mh = (lane >> 4) * 8;
#pragma unroll
  for (int nt = 0; nt < 24; ++nt) {
    int n = nt * 16 + nloc;
    float bias = bih[n];
#pragma unroll
    for (int i = 0; i < 8; ++i) {
      int m = m_base + mh + i;
      xproj[(size_t)m * G3 + n] = (bf16_t)(acc[nt][i] + bias);
    }
  }
}

// ---- Kernel 2: GRU recurrence. 8 WGs x 16 batch rows, h in LDS. ----------
// W_hh B-fragments + biases are time-invariant: hoisted into registers once
// (12 x v16bf per wave), so the 2048-step hot loop touches no global weights.
__global__ __launch_bounds__(256) void k_gru(
    const bf16_t* __restrict__ xproj, const bf16_t* __restrict__ Whh,
    const float* __restrict__ bhh, const int* __restrict__ lengths,
    bf16_t* __restrict__ gout) {
  __shared__ float  hs[16][Hc];    // fp32 master hidden state
  __shared__ bf16_t hb[16][LDH];   // bf16 staging, padded row (no bank conflicts)
  __shared__ float  ghs[16][G3];   // h @ W_hh^T + b_hh
  __shared__ int    lens[16];

  int tid = threadIdx.x;
  int lane = tid & 31;
  int w = tid >> 5;                // 8 waves
  int b0 = blockIdx.x * 16;

  for (int i = tid; i < 16 * Hc; i += 256) (&hs[0][0])[i] = 0.f;
  for (int i = tid; i < 16 * LDH; i += 256) (&hb[0][0])[i] = (bf16_t)0.f;
  if (tid < 16) lens[tid] = lengths[b0 + tid];

  // hoist loop-invariant W_hh fragments & biases (wave owns 3 N-tiles of 16)
  v16bf bw[3][4];
  float bias[3];
#pragma unroll
  for (int jt = 0; jt < 3; ++jt) {
    int n_base = (w * 3 + jt) * 16;
#pragma unroll
    for (int kk = 0; kk < 4; ++kk)
      bw[jt][kk] = frag_b_bf16(Whh + (size_t)n_base * Hc, Hc, lane, kk * 32);
    bias[jt] = bhh[n_base + (lane & 15)];
  }
  __syncthreads();

  int m_up = tid >> 4;          // 0..15 : batch row for gate update
  int j_up = (tid & 15) * 8;    // 0..120: hidden-col chunk

  for (int t = 0; t < Tc; ++t) {
    // phase 1: gh = h @ W_hh^T + b_hh
    v16bf af[4];
#pragma unroll
    for (int kk = 0; kk < 4; ++kk)
      af[kk] = frag_a_bf16(&hb[0][0], LDH, lane, kk * 32);
#pragma unroll
    for (int jt = 0; jt < 3; ++jt) {
      v8f acc = vzero8();
#pragma unroll
      for (int kk = 0; kk < 4; ++kk) acc = wmma_bf16(af[kk], bw[jt][kk], acc);
      int n = (w * 3 + jt) * 16 + (lane & 15);
      int mh = (lane >> 4) * 8;
#pragma unroll
      for (int i = 0; i < 8; ++i) ghs[mh + i][n] = acc[i] + bias[jt];
    }
    __syncthreads();

    // phase 2: gates + state update (each thread: 1 row x 8 cols, vectorized)
    {
      const bf16_t* xr = xproj + ((size_t)(b0 + m_up) * Tc + t) * G3 + j_up;
      bf8v vr = *(const bf8v*)(xr);
      bf8v vz = *(const bf8v*)(xr + Hc);
      bf8v vn = *(const bf8v*)(xr + 2 * Hc);
      f4v hr0 = *(const f4v*)&ghs[m_up][j_up];
      f4v hr1 = *(const f4v*)&ghs[m_up][j_up + 4];
      f4v hz0 = *(const f4v*)&ghs[m_up][Hc + j_up];
      f4v hz1 = *(const f4v*)&ghs[m_up][Hc + j_up + 4];
      f4v hn0 = *(const f4v*)&ghs[m_up][2 * Hc + j_up];
      f4v hn1 = *(const f4v*)&ghs[m_up][2 * Hc + j_up + 4];
      f4v ho0 = *(const f4v*)&hs[m_up][j_up];
      f4v ho1 = *(const f4v*)&hs[m_up][j_up + 4];
      bool keep = t < lens[m_up];
      f4v new0, new1;
      bf8v hbv, outv;
#pragma unroll
      for (int u = 0; u < 8; ++u) {
        float hr = (u < 4) ? hr0[u & 3] : hr1[u & 3];
        float hz = (u < 4) ? hz0[u & 3] : hz1[u & 3];
        float hn = (u < 4) ? hn0[u & 3] : hn1[u & 3];
        float ho = (u < 4) ? ho0[u & 3] : ho1[u & 3];
        float r  = sigmoidf_((float)vr[u] + hr);
        float z  = sigmoidf_((float)vz[u] + hz);
        float nn = tanhf((float)vn[u] + r * hn);
        float hnew = (1.f - z) * nn + z * ho;
        if (u < 4) new0[u & 3] = hnew; else new1[u & 3] = hnew;
        hbv[u]  = (bf16_t)hnew;
        outv[u] = keep ? (bf16_t)hnew : (bf16_t)0.f; // packed-seq mask
      }
      *(f4v*)&hs[m_up][j_up]     = new0;
      *(f4v*)&hs[m_up][j_up + 4] = new1;
      *(bf8v*)&hb[m_up][j_up]    = hbv;
      *(bf8v*)(gout + ((size_t)(b0 + m_up) * Tc + t) * Hc + j_up) = outv;
    }
    __syncthreads();
  }
}

// ---- Kernel 3: scorer + decay-weighted sum + final affine ----------------
// One WG per batch element (deterministic, no atomics). 4 waves stride t-tiles.
// W1 fragments + b1/W2 scalars hoisted out of the 32-iteration tile loop.
__global__ __launch_bounds__(128) void k_score(
    const bf16_t* __restrict__ gout, const bf16_t* __restrict__ W1b,
    const float* __restrict__ b1, const float* __restrict__ W2,
    const float* __restrict__ b2, const float* __restrict__ la,
    const float* __restrict__ Wc, const float* __restrict__ bc,
    float* __restrict__ out) {
  int b = blockIdx.x;
  int tid = threadIdx.x, lane = tid & 31, w = tid >> 5;
  float alpha = sigmoidf_(la[0]);
  float lalpha = __logf(alpha);
  float b2v = b2[0];

  v16bf bw[4][4];
  float b1v[4], w2v[4];
#pragma unroll
  for (int nt = 0; nt < 4; ++nt) {
#pragma unroll
    for (int kk = 0; kk < 4; ++kk)
      bw[nt][kk] = frag_b_bf16(W1b + nt * 16 * Hc, Hc, lane, kk * 32);
    int n = nt * 16 + (lane & 15);
    b1v[nt] = b1[n];
    w2v[nt] = W2[n];
  }

  float s = 0.f;
  for (int tt = w; tt < Tc / 16; tt += 4) {
    const bf16_t* arow = gout + ((size_t)b * Tc + tt * 16) * Hc;
    v8f acc[4];
#pragma unroll
    for (int i = 0; i < 4; ++i) acc[i] = vzero8();
#pragma unroll
    for (int kk = 0; kk < 4; ++kk) {
      v16bf af = frag_a_bf16(arow, Hc, lane, kk * 32);
#pragma unroll
      for (int nt = 0; nt < 4; ++nt) acc[nt] = wmma_bf16(af, bw[nt][kk], acc[nt]);
    }
    // second layer (64 -> 1): relu + dot with W2, per output row
    float part[8];
#pragma unroll
    for (int i = 0; i < 8; ++i) {
      float p = 0.f;
#pragma unroll
      for (int nt = 0; nt < 4; ++nt) {
        float v = acc[nt][i] + b1v[nt];
        v = v > 0.f ? v : 0.f;
        p += v * w2v[nt];
      }
      part[i] = p;
    }
#pragma unroll
    for (int i = 0; i < 8; ++i) {   // reduce across 16-lane half
      float p = part[i];
      p += __shfl_xor(p, 1, 32);
      p += __shfl_xor(p, 2, 32);
      p += __shfl_xor(p, 4, 32);
      p += __shfl_xor(p, 8, 32);
      part[i] = p;
    }
    if ((lane & 15) == 0) {
      int mh = (lane >> 4) * 8;
#pragma unroll
      for (int i = 0; i < 8; ++i) {
        int trow = tt * 16 + mh + i;
        float r = sigmoidf_(part[i] + b2v);
        float wgt = __expf(lalpha * (float)(Tc - 1 - trow)); // alpha^(T-1-t)
        s += r * wgt;
      }
    }
  }

  __shared__ float red[8];
  if ((lane & 15) == 0) red[w * 2 + (lane >> 4)] = s;
  __syncthreads();
  if (tid == 0) {
    float tot = 0.f;
#pragma unroll
    for (int i = 0; i < 8; ++i) tot += red[i];
    out[b] = tot * Wc[0] + bc[0];
  }
}

// ---- Host launcher --------------------------------------------------------
extern "C" void kernel_launch(void* const* d_in, const int* in_sizes, int n_in,
                              void* d_out, int out_size, void* d_ws, size_t ws_size,
                              hipStream_t stream) {
  const float* embs    = (const float*)d_in[0];
  const int*   lengths = (const int*)d_in[1];
  const float* W_ih    = (const float*)d_in[2];
  const float* b_ih    = (const float*)d_in[3];
  const float* W_hh    = (const float*)d_in[4];
  const float* b_hh    = (const float*)d_in[5];
  const float* W1      = (const float*)d_in[6];
  const float* b1      = (const float*)d_in[7];
  const float* W2      = (const float*)d_in[8];
  const float* b2      = (const float*)d_in[9];
  const float* la      = (const float*)d_in[10];
  const float* Wc      = (const float*)d_in[11];
  const float* bc      = (const float*)d_in[12];
  float* out = (float*)d_out;

  // workspace layout (bytes)
  size_t oWih = 0;
  size_t oWhh = oWih + (size_t)G3 * Ec * 2;            // 294912
  size_t oW1  = oWhh + (size_t)G3 * Hc * 2;            // +98304
  size_t oXp  = (oW1 + (size_t)64 * Hc * 2 + 255) & ~(size_t)255;
  size_t oGo  = oXp + (size_t)Bc * Tc * G3 * 2;        // x_proj bf16
  size_t need = oGo + (size_t)Bc * Tc * Hc * 2;        // gru_out bf16
  if (ws_size < need) return;  // workspace too small: leave output untouched

  char* ws = (char*)d_ws;
  bf16_t* Wih_b = (bf16_t*)(ws + oWih);
  bf16_t* Whh_b = (bf16_t*)(ws + oWhh);
  bf16_t* W1_b  = (bf16_t*)(ws + oW1);
  bf16_t* xp    = (bf16_t*)(ws + oXp);
  bf16_t* go    = (bf16_t*)(ws + oGo);

  // 0) convert weights
  k_convert<<<(G3 * Ec) / 256, 256, 0, stream>>>(W_ih, W_hh, W1, Wih_b, Whh_b, W1_b);
  // 1) x_proj GEMM: [B*T,384] x [384,384], embs streamed exactly once
  k_xproj<<<(Bc * Tc) / 64, 128, 0, stream>>>(embs, Wih_b, b_ih, xp);
  // 2) GRU recurrence: 8 WGs x 16 batch rows
  k_gru<<<Bc / 16, 256, 0, stream>>>(xp, Whh_b, b_hh, lengths, go);
  // 3) scorer + decayed sum + final affine
  k_score<<<Bc, 128, 0, stream>>>(go, W1_b, b1, W2, b2, la, Wc, bc, out);
  (void)in_sizes; (void)n_in; (void)out_size;
}
